// GroupQueryAttention_23622320128315
// MI455X (gfx1250) — matrix operand order repaired and checked by reading, not compiled, not run on hardware
//
#include <hip/hip_runtime.h>
#include <hip/hip_bf16.h>
#include <math.h>

// ---------------------------------------------------------------------------
// GQA forward on MI455X (gfx1250, wave32, WMMA).
// Precision: bf16 operands, fp32 accumulation (v_wmma_f32_16x16x32_bf16).
// GEMMs: block-cooperative 128x64 tiles, B panel double-buffered into LDS
// via global_load_async_to_lds_b128 (ASYNCcnt), A streamed with prefetch.
// ---------------------------------------------------------------------------

typedef __bf16 bf16;
typedef __attribute__((ext_vector_type(16))) __bf16 v16bf;
typedef __attribute__((ext_vector_type(8)))  __bf16 v8bf;
typedef __attribute__((ext_vector_type(8)))  float   v8f;

#define WMMA_BF16(a, b, c) \
  __builtin_amdgcn_wmma_f32_16x16x32_bf16(false, (a), false, (b), (short)0, (c), false, false)

constexpr int B_  = 2;
constexpr int S_  = 2048;
constexpr int D_  = 2048;
constexpr int H_  = 16;
constexpr int HK_ = 4;
constexpr int HD_ = 128;
constexpr int BS  = B_ * S_;                    // 4096 tokens
constexpr int NQKV = H_*HD_ + 2*HK_*HD_;        // 3072 = q(2048)|k(512)|v(512)
constexpr int KOFF = H_*HD_;                    // 2048: k section start
constexpr int VOFF = H_*HD_ + HK_*HD_;          // 2560: v section start
constexpr int KT   = 32;                        // keys per flash tile

// ---- A-fragment (16x32 bf16) loader per ISA layout ------------------------
// lane<16  : row = lane,    K in {base+0..7,  base+16..23}
// lane>=16 : row = lane-16, K in {base+8..15, base+24..31}
// caller passes p = &A[row*ld + kbase + half*8]
__device__ inline v16bf load_afrag(const bf16* p) {
  v8bf lo = *(const v8bf*)(p);
  v8bf hi = *(const v8bf*)(p + 16);
  v16bf a;
#pragma unroll
  for (int i = 0; i < 8; i++) { a[i] = lo[i]; a[i + 8] = hi[i]; }
  return a;
}

// ---------------------------------------------------------------------------
// Prep kernels: fp32 -> bf16
// ---------------------------------------------------------------------------
__global__ void cvt_f32_bf16(const float* __restrict__ src, bf16* __restrict__ dst, int n) {
  for (int i = blockIdx.x * blockDim.x + threadIdx.x; i < n; i += gridDim.x * blockDim.x)
    dst[i] = (bf16)src[i];
}

__global__ void pack_qkv_w(const float* __restrict__ wq, const float* __restrict__ wk,
                           const float* __restrict__ wv, bf16* __restrict__ dst) {
  const int n = D_ * NQKV;
  for (int i = blockIdx.x * blockDim.x + threadIdx.x; i < n; i += gridDim.x * blockDim.x) {
    int k = i / NQKV, c = i % NQKV;
    float val;
    if (c < KOFF)       val = wq[k * (H_*HD_)  + c];
    else if (c < VOFF)  val = wk[k * (HK_*HD_) + (c - KOFF)];
    else                val = wv[k * (HK_*HD_) + (c - VOFF)];
    dst[i] = (bf16)val;
  }
}

// ---------------------------------------------------------------------------
// WMMA GEMM:  C[M,N] = A[M,K] * B[K,N]   (A,B bf16 row-major, acc fp32)
// Block = 8 waves -> 128M x 64N tile. B panel (32x64 bf16 = 4KB) is shared by
// all waves and double-buffered into LDS with async copies; each wave reuses
// its A fragment across 4 WMMAs.
// ---------------------------------------------------------------------------
template <bool STORE_BF16>
__global__ __launch_bounds__(256)
void gemm_wmma(const bf16* __restrict__ A, const bf16* __restrict__ Bm,
               void* __restrict__ Cv, int M, int N, int K) {
  __shared__ alignas(32) bf16 ldsB[2][32 * 64];   // [buf][krow][ncol], 2 x 4KB

  const int tid  = threadIdx.x;
  const int lane = tid & 31;
  const int wv   = tid >> 5;
  const int r    = lane & 15;
  const int hf   = lane >> 4;

  const int Mb = M >> 7;                       // 128-row block tiles
  const int bm = blockIdx.x % Mb;
  const int bn = blockIdx.x / Mb;
  const int m0 = bm * 128 + wv * 16;
  const int n0 = bn * 64;

  // async staging: thread tid copies one 16B chunk of the 32x64 B tile
  const int srow = tid >> 3;                   // 0..31 (k row)
  const int scol = (tid & 7) * 8;              // 0..56 (n col, 8 bf16)
  const unsigned int ldsBase =
      (unsigned int)(unsigned long long)(&ldsB[0][0]);

#define STAGE_B(kk, buf)                                                       \
  do {                                                                         \
    unsigned long long ga =                                                    \
        (unsigned long long)(Bm + (size_t)((kk) + srow) * N + n0 + scol);      \
    unsigned int la = ldsBase + (buf) * 4096u + (srow * 64 + scol) * 2u;       \
    asm volatile("global_load_async_to_lds_b128 %0, %1, off"                   \
                 :: "v"(la), "v"(ga) : "memory");                              \
  } while (0)

  v8f acc[4];
#pragma unroll
  for (int t = 0; t < 4; t++) acc[t] = (v8f)0.0f;

  const bf16* arow = A + (size_t)(m0 + r) * K + hf * 8;

  STAGE_B(0, 0);                               // prologue
  for (int k0 = 0; k0 < K; k0 += 32) {
    const int cur = (k0 >> 5) & 1;
    if (k0 + 32 < K) {                         // stream next panel into alt buf
      STAGE_B(k0 + 32, cur ^ 1);
      asm volatile("s_wait_asynccnt 0x1" ::: "memory");
    } else {
      asm volatile("s_wait_asynccnt 0x0" ::: "memory");
    }
    __syncthreads();                           // cur panel visible block-wide

    __builtin_prefetch(arow + k0 + 64, 0, 3);  // A stream ahead
    v16bf a = load_afrag(arow + k0);
    const bf16* bp = &ldsB[cur][hf * 8 * 64 + r];    // lane&15 = N col, K in table order
#pragma unroll
    for (int t = 0; t < 4; t++) {
      v16bf b;
#pragma unroll
      for (int e = 0; e < 16; e++) b[e] = bp[((e & 7) + (e >> 3) * 16) * 64 + t * 16];
      acc[t] = WMMA_BF16(a, b, acc[t]);
    }

    asm volatile("s_wait_dscnt 0x0" ::: "memory");  // wave done reading cur
    __syncthreads();                           // all waves done -> alt reusable
  }
#undef STAGE_B

  // C layout: VGPR v -> row m0 + v + hf*8, lane&15 -> col
#pragma unroll
  for (int t = 0; t < 4; t++)
#pragma unroll
    for (int v = 0; v < 8; v++) {
      size_t idx = (size_t)(m0 + v + hf * 8) * N + (n0 + t * 16 + r);
      if (STORE_BF16) ((bf16*)Cv)[idx] = (bf16)acc[t][v];
      else            ((float*)Cv)[idx] = acc[t][v];
    }
}

// ---------------------------------------------------------------------------
// RoPE + RMSNorm (in place on bf16 qkv). One wave per (token, head-slot):
// slots 0..15 = q heads, 16..19 = k heads. 4 elems/lane over HD=128.
// ---------------------------------------------------------------------------
__global__ __launch_bounds__(256)
void rope_rms(bf16* __restrict__ qkv, const float* __restrict__ cosT,
              const float* __restrict__ sinT) {
  const int lane = threadIdx.x & 31;
  const int wv   = threadIdx.x >> 5;
  const int task = blockIdx.x * 8 + wv;        // [0, BS*20)
  const int t = task / (H_ + HK_);
  const int u = task % (H_ + HK_);
  const int s = t & (S_ - 1);
  const int col0 = (u < H_) ? u * HD_ : KOFF + (u - H_) * HD_;
  bf16* row = qkv + (size_t)t * NQKV + col0;
  const float* cr = cosT + s * HD_;
  const float* sr = sinT + s * HD_;

  float x0 = (float)row[lane],      x1 = (float)row[lane + 32];
  float x2 = (float)row[lane + 64], x3 = (float)row[lane + 96];
  float o0 = x0 * cr[lane]      - x2 * sr[lane];        // first half: x*c - x2*s
  float o1 = x1 * cr[lane + 32] - x3 * sr[lane + 32];
  float o2 = x2 * cr[lane + 64] + x0 * sr[lane + 64];   // second half: x*c + x1*s
  float o3 = x3 * cr[lane + 96] + x1 * sr[lane + 96];

  float ss = o0*o0 + o1*o1 + o2*o2 + o3*o3;
#pragma unroll
  for (int m = 1; m < 32; m <<= 1) ss += __shfl_xor(ss, m, 32);
  float rr = __frsqrt_rn(ss * (1.0f / HD_) + 1.1920929e-7f) * 1.2f;

  row[lane]      = (bf16)(o0 * rr);
  row[lane + 32] = (bf16)(o1 * rr);
  row[lane + 64] = (bf16)(o2 * rr);
  row[lane + 96] = (bf16)(o3 * rr);
}

// ---------------------------------------------------------------------------
// Flash attention: one wave per (b, head h, 16-query tile). Online softmax
// over 32-key tiles. K tile staged transposed in LDS -> QK^T B-frags are
// contiguous 32B LDS reads. P converted C-layout -> A-layout via 1KB LDS
// bounce. V B-frags load straight from global (lane&15 = d col, elements = key rows).
// ---------------------------------------------------------------------------
__global__ __launch_bounds__(128)
void flash_attn(const bf16* __restrict__ qkv, bf16* __restrict__ attnO) {
  __shared__ bf16 ldsKT[4][HD_ * KT];   // [d][key] transposed, 8KB/wave
  __shared__ bf16 ldsP[4][16 * KT];     // [q][key], 1KB/wave

  const int lane = threadIdx.x & 31;
  const int wv   = threadIdx.x >> 5;
  const int r    = lane & 15;
  const int hf   = lane >> 4;
  const int task = blockIdx.x * 4 + wv;        // [0, 4096)
  const int qt  = task & 127;
  const int h   = (task >> 7) & 15;
  const int b   = task >> 11;
  const int kvh = h >> 2;                      // h = kvh*R + r_sub
  const int q0  = qt * 16;
  const size_t rowb = (size_t)(b * S_);

  // Q fragments (loop invariant): 16 queries x 128 dims = 4 A-frags
  v16bf qa[4];
  {
    const bf16* qp = qkv + (rowb + q0 + r) * NQKV + h * HD_ + hf * 8;
#pragma unroll
    for (int dc = 0; dc < 4; dc++) qa[dc] = load_afrag(qp + dc * 32);
  }

  v8f o[8];
#pragma unroll
  for (int j = 0; j < 8; j++) o[j] = (v8f)0.0f;
  float m8[8], l8[8];
#pragma unroll
  for (int v = 0; v < 8; v++) { m8[v] = -__builtin_inff(); l8[v] = 0.0f; }

  const float sc = 0.08838834764831845f;   // 1/sqrt(128)
  const int ktiles = (q0 + 16 + KT - 1) / KT;
  bf16* kl = ldsKT[wv];
  bf16* pl = ldsP[wv];

  for (int kt = 0; kt < ktiles; kt++) {
    const int k0 = kt * KT;

    // -- stage K tile (32 keys x 128d) transposed into LDS ----------------
    asm volatile("s_wait_dscnt 0x0" ::: "memory");   // WAR vs prior tile's reads
#pragma unroll
    for (int it = 0; it < 16; it++) {
      int cid = it * 32 + lane;                  // 512 chunks of 8 bf16
      int key = cid >> 4, dch = cid & 15;
      v8bf kvv = *(const v8bf*)(qkv + (rowb + k0 + key) * NQKV + KOFF + kvh * HD_ + dch * 8);
#pragma unroll
      for (int j = 0; j < 8; j++) kl[(dch * 8 + j) * KT + key] = kvv[j];
    }
    asm volatile("s_wait_dscnt 0x0" ::: "memory");

    // -- S = Q * K^T : two 16x16 fragments (32 keys) ----------------------
    v8f s[2];
#pragma unroll
    for (int n = 0; n < 2; n++) {
      v8f a = (v8f)0.0f;
#pragma unroll
      for (int dc = 0; dc < 4; dc++) {
        v16bf bk;
#pragma unroll
        for (int e = 0; e < 16; e++) bk[e] = kl[(dc * 32 + hf * 8 + (e & 7) + (e >> 3) * 16) * KT + n * 16 + r];
        a = WMMA_BF16(qa[dc], bk, a);
      }
      s[n] = a;
    }

    // -- scale + causal mask ---------------------------------------------
#pragma unroll
    for (int n = 0; n < 2; n++) {
      int key = k0 + n * 16 + r;
#pragma unroll
      for (int v = 0; v < 8; v++) {
        float val = s[n][v] * sc;
        if (key > q0 + v + hf * 8) val = -__builtin_inff();
        s[n][v] = val;
      }
    }

    // -- online softmax: row stats live per-VGPR, reduce across 16 lanes --
    float rowm[8];
#pragma unroll
    for (int v = 0; v < 8; v++) rowm[v] = fmaxf(s[0][v], s[1][v]);
#pragma unroll
    for (int off = 1; off < 16; off <<= 1)
#pragma unroll
      for (int v = 0; v < 8; v++) rowm[v] = fmaxf(rowm[v], __shfl_xor(rowm[v], off, 32));

    float rscale[8];
#pragma unroll
    for (int v = 0; v < 8; v++) {
      float mn = fmaxf(m8[v], rowm[v]);
      rscale[v] = __expf(m8[v] - mn);      // exp(-inf)=0 on first tile
      m8[v] = mn;
    }
#pragma unroll
    for (int n = 0; n < 2; n++)
#pragma unroll
      for (int v = 0; v < 8; v++) s[n][v] = __expf(s[n][v] - m8[v]);

    float rsum[8];
#pragma unroll
    for (int v = 0; v < 8; v++) rsum[v] = s[0][v] + s[1][v];
#pragma unroll
    for (int off = 1; off < 16; off <<= 1)
#pragma unroll
      for (int v = 0; v < 8; v++) rsum[v] += __shfl_xor(rsum[v], off, 32);
#pragma unroll
    for (int v = 0; v < 8; v++) l8[v] = l8[v] * rscale[v] + rsum[v];
#pragma unroll
    for (int j = 0; j < 8; j++)
#pragma unroll
      for (int v = 0; v < 8; v++) o[j][v] *= rscale[v];

    // -- P: C-layout -> A-layout via LDS (bf16) ---------------------------
#pragma unroll
    for (int n = 0; n < 2; n++)
#pragma unroll
      for (int v = 0; v < 8; v++)
        pl[(v + hf * 8) * KT + n * 16 + r] = (bf16)s[n][v];
    asm volatile("s_wait_dscnt 0x0" ::: "memory");
    v16bf pa = load_afrag(pl + r * KT + hf * 8);

    // -- O += P * V (V B-frags direct from global: lane&15 = d col) --------
#pragma unroll
    for (int j = 0; j < 8; j++) {
      v16bf bv;
#pragma unroll
      for (int e = 0; e < 16; e++) bv[e] = qkv[(rowb + k0 + hf * 8 + (e & 7) + (e >> 3) * 16) * NQKV + VOFF + kvh * HD_ + j * 16 + r];
      o[j] = WMMA_BF16(pa, bv, o[j]);
    }
  }

  // epilogue: O /= l, store bf16 into attn buffer [BS, H*HD]
#pragma unroll
  for (int v = 0; v < 8; v++) {
    float inv = 1.0f / l8[v];
    size_t row = (rowb + q0 + v + hf * 8) * (size_t)(H_ * HD_);
#pragma unroll
    for (int j = 0; j < 8; j++)
      attnO[row + h * HD_ + j * 16 + r] = (bf16)(o[j][v] * inv);
  }
}

// ---------------------------------------------------------------------------
// Launch. Inputs: x, cos, sin, Wq, Wk, Wv, Wo (all fp32). Output fp32 [B,S,D].
// Workspace layout (bytes):
//   xb   @ 0         : 16 MB  bf16 [4096,2048]
//   wcv  @ 16777216  : 12 MB  bf16 [2048,3072]  (packed Wq|Wk|Wv)
//   wob  @ 29360128  :  8 MB  bf16 [2048,2048]
//   qkv  @ 37748736  : 24 MB  bf16 [4096,3072]
//   attn @ 62914560  : 16 MB  bf16 [4096,2048]        total ~76 MB
// ---------------------------------------------------------------------------
extern "C" void kernel_launch(void* const* d_in, const int* in_sizes, int n_in,
                              void* d_out, int out_size, void* d_ws, size_t ws_size,
                              hipStream_t stream) {
  const float* x    = (const float*)d_in[0];
  const float* cosT = (const float*)d_in[1];
  const float* sinT = (const float*)d_in[2];
  const float* Wq   = (const float*)d_in[3];
  const float* Wk   = (const float*)d_in[4];
  const float* Wv   = (const float*)d_in[5];
  const float* Wo   = (const float*)d_in[6];

  char* ws   = (char*)d_ws;
  bf16* xb   = (bf16*)(ws);
  bf16* wcv  = (bf16*)(ws + 16777216);
  bf16* wob  = (bf16*)(ws + 29360128);
  bf16* qkv  = (bf16*)(ws + 37748736);
  bf16* attn = (bf16*)(ws + 62914560);

  // 1) precision conversion / weight packing
  cvt_f32_bf16<<<4096, 256, 0, stream>>>(x,  xb,  BS * D_);
  pack_qkv_w  <<<4096, 256, 0, stream>>>(Wq, Wk, Wv, wcv);
  cvt_f32_bf16<<<4096, 256, 0, stream>>>(Wo, wob, D_ * D_);

  // 2) fused QKV projection: [4096,2048] x [2048,3072] -> bf16 qkv
  //    grid = (M/128) * (N/64) = 32 * 48
  gemm_wmma<true><<<1536, 256, 0, stream>>>(xb, wcv, qkv, BS, NQKV, D_);

  // 3) RoPE + RMSNorm in place (q and k sections)
  rope_rms<<<10240, 256, 0, stream>>>(qkv, cosT, sinT);

  // 4) flash attention -> bf16 attn [4096, 2048]
  flash_attn<<<1024, 128, 0, stream>>>(qkv, attn);

  // 5) output projection: [4096,2048] x [2048,2048] -> fp32 d_out
  //    grid = 32 * 32
  gemm_wmma<false><<<1024, 256, 0, stream>>>(attn, wob, d_out, BS, D_, D_);
}
